// LstmCrf_19241453486453
// MI455X (gfx1250) — matrix operand order repaired and checked by reading, not compiled, not run on hardware
//
#include <hip/hip_runtime.h>
#include <hip/hip_bf16.h>
#include <math.h>

// ---------------- problem constants ----------------
#define BSZ   64
#define SEQ   200
#define EMB   300
#define EMBP  320          // EMB padded to multiple of 32 (WMMA K granularity)
#define HID   512
#define G4    2048         // 4*HID gate dimension
#define KTOT  832          // HID + EMBP : fused recurrent+input GEMM K
#define NKT   26           // KTOT/32
#define NL    66           // N_LABELS
#define NLP   80           // padded to multiple of 16
#define TSTART 64
#define TSTOP  65

typedef __attribute__((ext_vector_type(16))) __bf16 v16bf;
typedef __attribute__((ext_vector_type(8)))  __bf16 v8bf;
typedef __attribute__((ext_vector_type(8)))  float  v8f;

// float -> bf16 round-to-nearest-even
__device__ __forceinline__ __bf16 f2bf(float f) {
    union { float f; unsigned u; } in; in.f = f;
    unsigned u = in.u;
    unsigned r = (u + 0x7FFFu + ((u >> 16) & 1u)) >> 16;
    union { unsigned short s; __bf16 h; } o; o.s = (unsigned short)r;
    return o.h;
}

__device__ __forceinline__ float sigmoidf(float x) { return 1.f / (1.f + expf(-x)); }

// concat two 128-bit LDS/global loads into one WMMA operand
__device__ __forceinline__ v16bf frag2(const __bf16* p) {
    v8bf lo = *(const v8bf*)p;
    v8bf hi = *(const v8bf*)(p + 8);
    return __builtin_shufflevector(lo, hi, 0,1,2,3,4,5,6,7,8,9,10,11,12,13,14,15);
}

// ---------------- weight pre-swizzle into WMMA fragment order ----------------
// LSTM B operand: BW[kt][jb][gate][lane][e], jb = hidden slice of 16, 32 slices.
//   k = kt*32 + (lane>>4)*8 + (e&7) + (e>>3)*16 ; col = jb*16 + (lane&15) ; gcol = gate*512+col
//   k <  512          -> W_hh[gcol*512 + k]
//   512 <= k < 812    -> W_ih[gcol*300 + (k-512)]
//   else zero pad
__global__ __launch_bounds__(256) void k_packB_lstm(const float* __restrict__ W_hh,
                                                    const float* __restrict__ W_ih,
                                                    __bf16* __restrict__ dst) {
    int idx = blockIdx.x * 256 + threadIdx.x;           // over NKT*32*4*32*16
    if (idx >= NKT * 32 * 4 * 32 * 16) return;
    int e    = idx & 15;
    int lane = (idx >> 4) & 31;
    int gate = (idx >> 9) & 3;
    int jb   = (idx >> 11) & 31;
    int kt   = idx >> 16;
    int k    = kt * 32 + (lane >> 4) * 8 + (e & 7) + (e >> 3) * 16;
    int col  = jb * 16 + (lane & 15);
    int gcol = gate * HID + col;
    float v;
    if (k < HID)              v = W_hh[(size_t)gcol * HID + k];
    else if (k - HID < EMB)   v = W_ih[(size_t)gcol * EMB + (k - HID)];
    else                      v = 0.f;
    dst[idx] = f2bf(v);
}
// feats B operand: BF[kt][q][lane][e]; n = q*16+(lane&15); k = kt*32+(lane>>4)*8+(e&7)+(e>>3)*16
__global__ __launch_bounds__(256) void k_packB_fc(const float* __restrict__ W_fc,
                                                  __bf16* __restrict__ dst) {
    int idx = blockIdx.x * 256 + threadIdx.x;           // over 16*5*32*16
    if (idx >= 16 * 5 * 32 * 16) return;
    int e    = idx & 15;
    int lane = (idx >> 4) & 31;
    int r    = idx >> 9;
    int q    = r % 5;
    int kt   = r / 5;
    int k    = kt * 32 + (lane >> 4) * 8 + (e & 7) + (e >> 3) * 16;
    int n    = q * 16 + (lane & 15);
    float v  = (n < NL) ? W_fc[(size_t)n * HID + k] : 0.f;
    dst[idx] = f2bf(v);
}
// zero h0 (bf16) and c (f32)
__global__ __launch_bounds__(256) void k_zero(__bf16* __restrict__ h0,
                                              float* __restrict__ c) {
    int idx = blockIdx.x * 256 + threadIdx.x;
    if (idx >= BSZ * HID) return;
    h0[idx] = f2bf(0.f);
    c[idx]  = 0.f;
}

// ---------------- embedding: gather + max-norm renorm + bf16 ----------------
__global__ __launch_bounds__(256) void k_embed(const int* __restrict__ data,
                                               const float* __restrict__ table,
                                               __bf16* __restrict__ out) {
    int tid = threadIdx.x, lane = tid & 31, wv = tid >> 5;
    int tok = blockIdx.x * 8 + wv;
    if (tok >= BSZ * SEQ) return;
    const float* row = table + (size_t)data[tok] * EMB;
    float ss = 0.f;
    for (int e = lane; e < EMB; e += 32) { float v = row[e]; ss += v * v; }
    #pragma unroll
    for (int m = 16; m > 0; m >>= 1) ss += __shfl_xor(ss, m, 32);
    float nrm   = sqrtf(ss);
    float scale = fminf(1.f, 6.0f / fmaxf(nrm, 1e-7f));
    __bf16* dst = out + (size_t)tok * EMBP;
    for (int e = lane; e < EMBP; e += 32)
        dst[e] = f2bf((e < EMB) ? row[e] * scale : 0.f);
}

// ---------------- fused LSTM step: gates = [h | x_t] @ [W_hh;W_ih]^T, K=832 ----------------
// 32 blocks x 64 threads (2 waves). Block = hidden slice of 16 (j0 = bx*16);
// N-tiles = 4 gates over the same 16 hidden cols -> i/f/g/o co-resident per wave.
// Wave w handles M-tiles {w, w+2}; B fragments shared across both M-tiles.
// No LDS: A fragments are two contiguous 16B runs in row-major h/emb; B fragments
// come pre-swizzled from global (L2-resident bf16 weights).
__global__ __launch_bounds__(64) void k_lstm_step(
        const __bf16* __restrict__ h_in, __bf16* __restrict__ h_out,
        float* __restrict__ c_state,     const __bf16* __restrict__ emb,
        const __bf16* __restrict__ BW,   const float* __restrict__ bias,
        __bf16* __restrict__ hs, int t) {
    int tid = threadIdx.x, lane = tid & 31, wave = tid >> 5;
    int jb = blockIdx.x;                 // hidden slice
    int hi = lane >> 4, l15 = lane & 15;

    int mt0 = wave, mt1 = wave + 2;      // two M-tiles per wave
    int arow0 = mt0 * 16 + l15;
    int arow1 = mt1 * 16 + l15;
    const __bf16* ah0 = h_in + (size_t)arow0 * HID;
    const __bf16* ah1 = h_in + (size_t)arow1 * HID;
    const __bf16* ae0 = emb + ((size_t)arow0 * SEQ + t) * EMBP - HID;
    const __bf16* ae1 = emb + ((size_t)arow1 * SEQ + t) * EMBP - HID;
    const __bf16* bbase = BW + (((size_t)jb * 4) * 32) * 16;   // + kt*32*4*32*16 later

    v8f acc[2][4];
    #pragma unroll
    for (int s = 0; s < 2; ++s)
        #pragma unroll
        for (int q = 0; q < 4; ++q)
            #pragma unroll
            for (int e = 0; e < 8; ++e) acc[s][q][e] = 0.f;

    for (int kt = 0; kt < NKT; ++kt) {
        int kbase = kt * 32;
        const __bf16* a0 = (kbase < HID) ? ah0 + kbase : ae0 + kbase;
        const __bf16* a1 = (kbase < HID) ? ah1 + kbase : ae1 + kbase;
        // A fragment: rows in lanes 0-15, K = [hi*8,+8) and [16+hi*8,+8) contiguous
        v16bf af0, af1;
        {
            v8bf x0 = *(const v8bf*)(a0 + hi * 8);
            v8bf x1 = *(const v8bf*)(a0 + 16 + hi * 8);
            af0 = __builtin_shufflevector(x0, x1, 0,1,2,3,4,5,6,7,8,9,10,11,12,13,14,15);
            v8bf y0 = *(const v8bf*)(a1 + hi * 8);
            v8bf y1 = *(const v8bf*)(a1 + 16 + hi * 8);
            af1 = __builtin_shufflevector(y0, y1, 0,1,2,3,4,5,6,7,8,9,10,11,12,13,14,15);
        }
        const __bf16* bkt = bbase + (size_t)kt * (32 * 4 * 32 * 16);
        #pragma unroll
        for (int q = 0; q < 4; ++q) {    // gates i,f,g,o
            v16bf bf = frag2(bkt + ((size_t)q * 32 + lane) * 16);
            acc[0][q] = __builtin_amdgcn_wmma_f32_16x16x32_bf16(
                false, af0, false, bf, (short)0, acc[0][q], false, false);
            acc[1][q] = __builtin_amdgcn_wmma_f32_16x16x32_bf16(
                false, af1, false, bf, (short)0, acc[1][q], false, false);
        }
    }

    // elementwise LSTM cell update, entirely in fragment registers
    int col = jb * 16 + l15;             // hidden unit (same for all accs of this lane)
    float bi = bias[0 * HID + col];
    float bf_ = bias[1 * HID + col];
    float bg = bias[2 * HID + col];
    float bo = bias[3 * HID + col];
    #pragma unroll
    for (int s = 0; s < 2; ++s) {
        int mt = wave + s * 2;
        #pragma unroll
        for (int v = 0; v < 8; ++v) {
            int m = mt * 16 + hi * 8 + v;            // batch row
            float iv = acc[s][0][v] + bi;
            float fv = acc[s][1][v] + bf_;
            float gv = acc[s][2][v] + bg;
            float ov = acc[s][3][v] + bo;
            float c_old = c_state[(size_t)m * HID + col];
            float cn = sigmoidf(fv) * c_old + sigmoidf(iv) * tanhf(gv);
            float hn = sigmoidf(ov) * tanhf(cn);
            c_state[(size_t)m * HID + col] = cn;
            __bf16 hb = f2bf(hn);
            h_out[(size_t)m * HID + col] = hb;
            hs[((size_t)m * SEQ + t) * HID + col] = hb;
        }
    }
}

// ---------------- feats = hs @ W_fc^T + b_fc  (M=12800, N=80 pad, K=512) ----------------
// 200 blocks x 128 threads (4 waves); wave = M-tile, 5 N-tiles. No LDS.
__global__ __launch_bounds__(128) void k_feats(const __bf16* __restrict__ hs,
                                               const __bf16* __restrict__ BF,
                                               const float*  __restrict__ b_fc,
                                               float* __restrict__ feats) {
    int tid = threadIdx.x, lane = tid & 31, wave = tid >> 5;
    int hi = lane >> 4, l15 = lane & 15;
    int m0 = blockIdx.x * 64;
    int arow = m0 + wave * 16 + l15;
    const __bf16* abase = hs + (size_t)arow * HID;

    v8f acc[5];
    #pragma unroll
    for (int q = 0; q < 5; ++q)
        #pragma unroll
        for (int e = 0; e < 8; ++e) acc[q][e] = 0.f;

    for (int kt = 0; kt < HID / 32; ++kt) {
        const __bf16* a = abase + kt * 32;
        v8bf x0 = *(const v8bf*)(a + hi * 8);
        v8bf x1 = *(const v8bf*)(a + 16 + hi * 8);
        v16bf af = __builtin_shufflevector(x0, x1, 0,1,2,3,4,5,6,7,8,9,10,11,12,13,14,15);
        #pragma unroll
        for (int q = 0; q < 5; ++q) {
            v16bf bf = frag2(BF + (((size_t)kt * 5 + q) * 32 + lane) * 16);
            acc[q] = __builtin_amdgcn_wmma_f32_16x16x32_bf16(
                false, af, false, bf, (short)0, acc[q], false, false);
        }
    }
    #pragma unroll
    for (int q = 0; q < 5; ++q) {
        int n = q * 16 + l15;
        if (n < NL) {
            float bn = b_fc[n];
            #pragma unroll
            for (int v = 0; v < 8; ++v) {
                int m = wave * 16 + hi * 8 + v;
                feats[(size_t)(m0 + m) * NL + n] = acc[q][v] + bn;
            }
        }
    }
}

// ---------------- CRF forward scan (log-partition) ----------------
__global__ __launch_bounds__(128) void k_crf(const float* __restrict__ feats,
                                             const int*   __restrict__ lengths,
                                             const float* __restrict__ trans,
                                             float* __restrict__ norm) {
    __shared__ float tr[NL * NL];
    __shared__ float alpha[2][NL];
    int b = blockIdx.x, tid = threadIdx.x;
    for (int i = tid; i < NL * NL; i += 128) tr[i] = trans[i];
    if (tid < NL) alpha[0][tid] = (tid == TSTART) ? 0.f : -10000.f;
    __syncthreads();
    int len = lengths[b];
    int cur = 0;
    for (int t = 0; t < SEQ; ++t) {
        if (tid < NL) {
            float mx = -1e30f;
            for (int j = 0; j < NL; ++j)
                mx = fmaxf(mx, tr[tid * NL + j] + alpha[cur][j]);
            float s = 0.f;
            for (int j = 0; j < NL; ++j)
                s += expf(tr[tid * NL + j] + alpha[cur][j] - mx);
            float an = mx + logf(s) + feats[((size_t)b * SEQ + t) * NL + tid];
            alpha[cur ^ 1][tid] = (t < len) ? an : alpha[cur][tid];
        }
        __syncthreads();
        cur ^= 1;
    }
    if (tid == 0) {
        float mx = -1e30f;
        for (int j = 0; j < NL; ++j)
            mx = fmaxf(mx, alpha[cur][j] + tr[TSTOP * NL + j]);
        float s = 0.f;
        for (int j = 0; j < NL; ++j)
            s += expf(alpha[cur][j] + tr[TSTOP * NL + j] - mx);
        norm[b] = mx + logf(s);
    }
}

// ---------------- gold score + final output ----------------
__global__ __launch_bounds__(64) void k_score(const float* __restrict__ feats,
                                              const int*   __restrict__ labels,
                                              const int*   __restrict__ lengths,
                                              const float* __restrict__ trans,
                                              const float* __restrict__ norm,
                                              float* __restrict__ out) {
    int b = blockIdx.x, tid = threadIdx.x;
    int len = lengths[b];
    float acc = 0.f;
    for (int t = tid; t < SEQ; t += 64)
        if (t < len)
            acc += feats[((size_t)b * SEQ + t) * NL + labels[b * SEQ + t]];
    for (int p = tid; p <= SEQ; p += 64) {
        if (p <= len) {
            int e0 = (p == 0) ? TSTART : labels[b * SEQ + p - 1];
            int q = p + 1;
            int e1 = (q <= len) ? labels[b * SEQ + q - 1] : TSTOP;
            acc += trans[e1 * NL + e0];
        }
    }
    #pragma unroll
    for (int m = 16; m > 0; m >>= 1) acc += __shfl_xor(acc, m, 32);
    __shared__ float part[2];
    if ((tid & 31) == 0) part[tid >> 5] = acc;
    __syncthreads();
    if (tid == 0) out[b] = norm[b] - (part[0] + part[1]);
}

// ---------------- host launch ----------------
extern "C" void kernel_launch(void* const* d_in, const int* in_sizes, int n_in,
                              void* d_out, int out_size, void* d_ws, size_t ws_size,
                              hipStream_t stream) {
    const int*   data      = (const int*)  d_in[0];
    const int*   lengths   = (const int*)  d_in[1];
    const int*   labels    = (const int*)  d_in[2];
    const float* emb_table = (const float*)d_in[3];
    const float* W_ih      = (const float*)d_in[4];
    const float* W_hh      = (const float*)d_in[5];
    const float* bias      = (const float*)d_in[6];
    const float* W_fc      = (const float*)d_in[7];
    const float* b_fc      = (const float*)d_in[8];
    const float* trans     = (const float*)d_in[9];
    float* out = (float*)d_out;

    char* base = (char*)d_ws;
    size_t off = 0;
    auto carve = [&](size_t bytes) { char* p = base + off; off = (off + bytes + 255) & ~(size_t)255; return p; };
    __bf16* emb_bf = (__bf16*)carve((size_t)BSZ * SEQ * EMBP * 2);
    __bf16* BW     = (__bf16*)carve((size_t)NKT * 32 * 4 * 32 * 16 * 2);  // 3.4 MB swizzled lstm weights
    __bf16* BF     = (__bf16*)carve((size_t)16 * 5 * 32 * 16 * 2);        // 80 KB swizzled fc weights
    __bf16* h0     = (__bf16*)carve((size_t)BSZ * HID * 2);
    __bf16* h1     = (__bf16*)carve((size_t)BSZ * HID * 2);
    float*  cst    = (float*) carve((size_t)BSZ * HID * 4);
    __bf16* hs     = (__bf16*)carve((size_t)BSZ * SEQ * HID * 2);
    float*  feats  = (float*) carve((size_t)BSZ * SEQ * NL * 4);
    float*  norm   = (float*) carve((size_t)BSZ * 4);

    // weight pre-swizzle + state init + embedding
    int nBW = NKT * 32 * 4 * 32 * 16;
    k_packB_lstm<<<(nBW + 255) / 256, 256, 0, stream>>>(W_hh, W_ih, BW);
    k_packB_fc<<<(16 * 5 * 32 * 16 + 255) / 256, 256, 0, stream>>>(W_fc, BF);
    k_zero<<<(BSZ * HID + 255) / 256, 256, 0, stream>>>(h0, cst);
    k_embed<<<(BSZ * SEQ) / 8, 256, 0, stream>>>(data, emb_table, emb_bf);

    // sequential recurrence: 200 dependent launches, ping-pong h buffers
    for (int t = 0; t < SEQ; ++t) {
        __bf16* hin  = (t & 1) ? h1 : h0;
        __bf16* hout = (t & 1) ? h0 : h1;
        k_lstm_step<<<32, 64, 0, stream>>>(hin, hout, cst, emb_bf, BW, bias, hs, t);
    }

    k_feats<<<(BSZ * SEQ) / 64, 128, 0, stream>>>(hs, BF, b_fc, feats);
    k_crf<<<BSZ, 128, 0, stream>>>(feats, lengths, trans, norm);
    k_score<<<BSZ, 64, 0, stream>>>(feats, labels, lengths, trans, norm, out);
}
